// Decoder_1726576855881
// MI455X (gfx1250) — compile-verified, matrix operand order checked by reading, not run on hardware
//
#include <hip/hip_runtime.h>

// ---------------- problem constants ----------------
#define NB   64
#define CCH  256
#define TT   60
#define VV   21
#define LATD 64
#define MTOT (NB * TT * VV)   // 80640
#define NTT  (NB * TT)        // 3840
#define LPAD 40               // LDS row pitch in halves (80B: 16B-aligned, bank-spread)

typedef __attribute__((ext_vector_type(16))) _Float16 v16h;
typedef __attribute__((ext_vector_type(8)))  float    v8f;

// joint permutation: part-grouped order (IDX flattened).  cPerm[vp] = original joint.
__constant__ int cPerm[VV]   = {1,2,3,4, 5,6,7,8, 0,9,10,11,12, 13,14,15,16, 17,18,19,20};
__constant__ int cPartOf[VV] = {0,0,0,0, 1,1,1,1, 2,2,2,2,2, 3,3,3,3, 4,4,4,4};

// ---------------- WMMA fragment helpers (ISA 7.12.2, 16-bit 16x32 A layout) -------
__device__ __forceinline__ int kmap(int lane, int h) {
  return ((h < 8) ? h : h + 8) + ((lane & 16) ? 8 : 0);
}
__device__ __forceinline__ v16h load_frag(const _Float16* row, int lane) {
  v16h f;
#pragma unroll
  for (int h = 0; h < 16; ++h) f[h] = row[kmap(lane, h)];
  return f;
}

// ---------------- generic WMMA GEMM:  out[o,m] = sum_k W[o,k] * B[k,m] ------------
// AMODE  0: W row-major OxK (f32)     1: tcn  W[(o*256+c)*3+dt], k = dt*256+c
// BMODE  2: f16 CM activations        3: tcn reflect-shifted f16 CM (plen==VV)
// OUTMODE 0: f32 CM   1: f16 CM   2: NCTV d_out (un-permute joints)
// Tiling is exact: Mdim % 128 == 0, Odim % 64 == 0, Kdim % 32 == 0 (no guards).
struct GemmArgs {
  const float*    W;
  const float*    bias;   // or nullptr
  const _Float16* B;
  const float*    resid;  // f32 CM at global m, or nullptr
  float*          out32;
  _Float16*       out16;
  float*          outNC;
  int Odim, Kdim, Mdim;
  int offs, plen;         // joint-range map (full tensor: offs=0, plen=VV)
};

template <int AMODE, int BMODE, int OUTMODE>
__global__ __launch_bounds__(256) void gemm_wmma_kernel(GemmArgs a) {
  __shared__ _Float16 As[64 * LPAD];
  __shared__ _Float16 Bs[128 * LPAD];
  const int tid = threadIdx.x, lane = tid & 31, wave = tid >> 5;
  const int oBase = blockIdx.y * 64, mBase = blockIdx.x * 128;
  const int waveO = wave >> 2, waveM = wave & 3;   // 2 x 4 waves, each 32(o) x 32(m)

  // --- per-thread B slot: one column, 16 contiguous k (2x b128 LDS store) ---
  const int colL  = tid & 127;
  const int kSub  = (tid >> 7) * 16;               // 0 or 16
  const int m     = mBase + colL;
  const int tl    = m / a.plen, j = m - tl * a.plen;  // tl = n*TT + t
  const int n     = tl / TT,   t = tl - n * TT;
  const int gm    = tl * VV + a.offs + j;

  v8f acc[2][2] = {};
  for (int k0 = 0; k0 < a.Kdim; k0 += 32) {
    // ---- A tile 64x32: lanes walk k (contiguous in W) ----
    float av[8];
#pragma unroll
    for (int i = 0; i < 8; ++i) {
      int row = oBase + wave + 8 * i;
      if (AMODE == 0) {
        av[i] = a.W[row * a.Kdim + k0 + lane];
      } else {
        int cc = (k0 & 255) + lane, dt = k0 >> 8;
        av[i] = a.W[(row * CCH + cc) * 3 + dt];
      }
    }
    // ---- B tile 128x32: lanes walk columns (contiguous in [c][m] layout) ----
    const _Float16* bp;
    if (BMODE == 2) {
      bp = a.B + (k0 + kSub) * MTOT + gm;
    } else {  // tcn: dt uniform per 32-slab (256 % 32 == 0)
      int dt = k0 >> 8;
      int ts = t + dt - 1;
      ts = (ts < 0) ? 1 : ((ts >= TT) ? TT - 2 : ts);
      bp = a.B + ((k0 & 255) + kSub) * MTOT + (n * TT + ts) * VV + j;
    }
    _Float16 bv[16];
#pragma unroll
    for (int i = 0; i < 16; ++i) bv[i] = bp[i * MTOT];
    if (k0 + 32 < a.Kdim) __builtin_prefetch(bp + 32 * MTOT, 0, 0);  // global_prefetch_b8

#pragma unroll
    for (int i = 0; i < 8; ++i) As[(wave + 8 * i) * LPAD + lane] = (_Float16)av[i];
#pragma unroll
    for (int i = 0; i < 16; ++i) Bs[colL * LPAD + kSub + i] = bv[i];
    __syncthreads();
#pragma unroll
    for (int oi = 0; oi < 2; ++oi) {
      v16h af = load_frag(&As[(waveO * 32 + oi * 16 + (lane & 15)) * LPAD], lane);
#pragma unroll
      for (int mi = 0; mi < 2; ++mi) {
        v16h bf = load_frag(&Bs[(waveM * 32 + mi * 16 + (lane & 15)) * LPAD], lane);
        acc[oi][mi] = __builtin_amdgcn_wmma_f32_16x16x32_f16(
            false, af, false, bf, (short)0, acc[oi][mi], false, false);
      }
    }
    __syncthreads();
  }
  // ---- epilogue (exact tiles, no guards; 2 divisions per thread) ----
  const int rowOff = (lane & 16) ? 8 : 0;
  const int colOff = lane & 15;
#pragma unroll
  for (int mi = 0; mi < 2; ++mi) {
    int mcol = mBase + waveM * 32 + mi * 16 + colOff;
    int tl2 = mcol / a.plen, j2 = mcol - tl2 * a.plen;
    int g2 = tl2 * VV + a.offs + j2;
#pragma unroll
    for (int oi = 0; oi < 2; ++oi)
#pragma unroll
      for (int r = 0; r < 8; ++r) {
        int orow = oBase + waveO * 32 + oi * 16 + rowOff + r;
        float v = acc[oi][mi][r];
        if (a.bias)  v += a.bias[orow];
        if (a.resid) v += a.resid[orow * MTOT + g2];
        if (OUTMODE == 0)      a.out32[orow * MTOT + g2] = v;
        else if (OUTMODE == 1) a.out16[orow * MTOT + g2] = (_Float16)v;
        else {
          int vp = g2 % VV, ntl = g2 / VV, t2 = ntl % TT, n2 = ntl / TT;
          a.outNC[((n2 * CCH + orow) * TT + t2) * VV + cPerm[vp]] = v;
        }
      }
  }
}

// ---------------- per-part instance-norm statistics -------------------------------
__constant__ int cOffs[5] = {0, 4, 8, 13, 17};
__constant__ int cLen[5]  = {4, 4, 5, 4, 4};

__global__ void stats_nctv_kernel(const float* src, int p, float* meanO, float* rstdO) {
  int n = blockIdx.x, c = threadIdx.x;
  int len = cLen[p], off = cOffs[p];
  float s = 0.f, s2 = 0.f;
  for (int t = 0; t < TT; ++t)
    for (int j = 0; j < len; ++j) {
      float v = src[((n * CCH + c) * TT + t) * VV + cPerm[off + j]];
      s += v; s2 += v * v;
    }
  float inv = 1.f / (float)(TT * len);
  float m = s * inv, var = s2 * inv - m * m;
  int si = (p * NB + n) * CCH + c;
  meanO[si] = m;
  rstdO[si] = rsqrtf(var + 1e-5f);
}

__global__ void stats_cm_kernel(const float* src, int p, float* meanO, float* rstdO) {
  int n = blockIdx.x, c = threadIdx.x;
  int len = cLen[p], off = cOffs[p];
  float s = 0.f, s2 = 0.f;
  for (int t = 0; t < TT; ++t)
    for (int j = 0; j < len; ++j) {
      float v = src[c * MTOT + (n * TT + t) * VV + off + j];
      s += v; s2 += v * v;
    }
  float inv = 1.f / (float)(TT * len);
  float m = s * inv, var = s2 * inv - m * m;
  int si = (p * NB + n) * CCH + c;
  meanO[si] = m;
  rstdO[si] = rsqrtf(var + 1e-5f);
}

// ---------------- AdaIN MLP: gamma/beta from style means --------------------------
__global__ void adain_mlp_kernel(const float* sMean, const float* wl, const float* bl,
                                 const float* w1, const float* b1,
                                 const float* w2, const float* b2, float* gbeta) {
  int p = blockIdx.y, n = blockIdx.x, tid = threadIdx.x;
  __shared__ float sm[CCH], z[LATD], h1[LATD];
  sm[tid] = sMean[(p * NB + n) * CCH + tid];
  __syncthreads();
  if (tid < LATD) {
    float a = bl[p * LATD + tid];
    const float* wr = &wl[(p * LATD + tid) * CCH];
    for (int c = 0; c < CCH; ++c) a += sm[c] * wr[c];
    z[tid] = (a > 0.f) ? a : 0.2f * a;
  }
  __syncthreads();
  if (tid < LATD) {
    float a = b1[p * LATD + tid];
    const float* wr = &w1[(p * LATD + tid) * LATD];
    for (int i = 0; i < LATD; ++i) a += z[i] * wr[i];
    h1[tid] = (a > 0.f) ? a : 0.2f * a;
  }
  __syncthreads();
  for (int o = tid; o < 2 * CCH; o += 256) {
    float a = b2[p * 2 * CCH + o];
    const float* wr = &w2[(p * 2 * CCH + o) * LATD];
    for (int i = 0; i < LATD; ++i) a += h1[i] * wr[i];
    gbeta[(p * NB + n) * 2 * CCH + o] = a;
  }
}

// ---------------- AdaIN apply + lrelu -> f16 CM (part-permuted) -------------------
__global__ void adain_apply_kernel(const float* x, const float* xMean, const float* xRstd,
                                   const float* gbeta, _Float16* act1h) {
  int m = blockIdx.x, c = threadIdx.x;
  int vp = m % VV, tl = m / VV, t = tl % TT, n = tl / TT;
  int p = cPartOf[vp];
  float xv = x[((n * CCH + c) * TT + t) * VV + cPerm[vp]];
  int si = (p * NB + n) * CCH + c;
  float g  = gbeta[(p * NB + n) * 2 * CCH + c];
  float be = gbeta[(p * NB + n) * 2 * CCH + CCH + c];
  float v = (1.f + g) * (xv - xMean[si]) * xRstd[si] + be;
  act1h[c * MTOT + m] = (_Float16)((v > 0.f) ? v : 0.2f * v);
}

// ---------------- pack styles: raw + normalized f16 CM ----------------------------
__global__ void pack_style_kernel(const float* s0, const float* s1, const float* s2,
                                  const float* s3, const float* s4,
                                  const float* sMean, const float* sRstd,
                                  _Float16* Sn, _Float16* Sr) {
  int m = blockIdx.x, c = threadIdx.x;
  int vp = m % VV, tl = m / VV, t = tl % TT, n = tl / TT;
  int p = cPartOf[vp];
  const float* sp = (p == 0) ? s0 : (p == 1) ? s1 : (p == 2) ? s2 : (p == 3) ? s3 : s4;
  float v = sp[((n * CCH + c) * TT + t) * VV + cPerm[vp]];
  int si = (p * NB + n) * CCH + c;
  Sr[c * MTOT + m] = (_Float16)v;
  Sn[c * MTOT + m] = (_Float16)((v - sMean[si]) * sRstd[si]);
}

// ---------------- normalize f32 CM -> f16 CM --------------------------------------
__global__ void normalize_cm_kernel(const float* src, const float* mean, const float* rstd,
                                    _Float16* dst) {
  int m = blockIdx.x, c = threadIdx.x;
  int vp = m % VV, tl = m / VV, n = tl / TT;
  int p = cPartOf[vp];
  int si = (p * NB + n) * CCH + c;
  dst[c * MTOT + m] = (_Float16)((src[c * MTOT + m] - mean[si]) * rstd[si]);
}

// ---------------- adjacency contraction (LDS-staged) ------------------------------
__global__ __launch_bounds__(256) void acontract_kernel(const _Float16* Y, const float* Amat,
                                                        const float* bias768, _Float16* out) {
  int nt = blockIdx.x, c = threadIdx.x;
  __shared__ float Ap[3][VV][VV];
  __shared__ float Asum[3][VV];
  __shared__ _Float16 Yl[3 * CCH * VV];
  for (int i = threadIdx.x; i < 3 * VV * VV; i += 256) {
    int k = i / (VV * VV), r = i % (VV * VV), vp = r / VV, wp = r % VV;
    Ap[k][vp][wp] = Amat[k * VV * VV + cPerm[vp] * VV + cPerm[wp]];
  }
  for (int i = threadIdx.x; i < 3 * CCH * VV; i += 256)
    Yl[i] = Y[(i / VV) * MTOT + nt * VV + (i % VV)];
  __syncthreads();
  if (threadIdx.x < 3 * VV) {
    int k = threadIdx.x / VV, wp = threadIdx.x % VV;
    float s = 0.f;
    for (int vp = 0; vp < VV; ++vp) s += Ap[k][vp][wp];
    Asum[k][wp] = s;
  }
  __syncthreads();
  float wacc[VV];
#pragma unroll
  for (int wp = 0; wp < VV; ++wp) wacc[wp] = 0.f;
  for (int k = 0; k < 3; ++k) {
    float bsc = bias768[k * CCH + c];
    float yrow[VV];
#pragma unroll
    for (int vp = 0; vp < VV; ++vp) yrow[vp] = (float)Yl[(k * CCH + c) * VV + vp];
#pragma unroll
    for (int wp = 0; wp < VV; ++wp) {
      float a2 = bsc * Asum[k][wp];
#pragma unroll
      for (int vp = 0; vp < VV; ++vp) a2 += yrow[vp] * Ap[k][vp][wp];
      wacc[wp] += a2;
    }
  }
#pragma unroll
  for (int wp = 0; wp < VV; ++wp) out[c * MTOT + nt * VV + wp] = (_Float16)wacc[wp];
}

// ---------------- attention scores: S[l,m] = sum_c F[c,l]*G[c,m] ------------------
// S has pitch Lpd (64-aligned). Pad rows/cols are written as ZEROS (consumed as the
// zero-padding of attn_out's K axis), so staging loads are unguarded (clamped index).
// All S indexing is 32-bit off a block-uniform base (N*Lpd^2 < 2^31).
__global__ __launch_bounds__(256) void attn_scores_kernel(const _Float16* F, const _Float16* G,
                                                          float* S, int Lp, int Lpd,
                                                          int plen, int offsj) {
  const int n = blockIdx.z;
  const int lBase = blockIdx.y * 64, mBase = blockIdx.x * 64;
  float* Sblk = S + (n * Lpd + lBase) * Lpd;       // block-uniform base
  __shared__ _Float16 Fs[64 * LPAD];
  __shared__ _Float16 Gs[64 * LPAD];
  const int tid = threadIdx.x, lane = tid & 31, wave = tid >> 5;
  const int waveL = wave >> 2, waveM = wave & 3;   // wave: 32(l) x 16(m)
  // per-thread slot: one row, 8 contiguous k (lanes walk rows = contiguous m axis)
  const int rloc  = tid & 63;
  const int kbase = (tid >> 6) * 8;
  int lG = lBase + rloc; if (lG >= Lp) lG = Lp - 1;   // clamp: garbage rows zeroed at store
  int mG = mBase + rloc; if (mG >= Lp) mG = Lp - 1;
  int gmL, gmM;
  { int t = lG / plen, j = lG - t * plen; gmL = (n * TT + t) * VV + offsj + j; }
  { int t = mG / plen, j = mG - t * plen; gmM = (n * TT + t) * VV + offsj + j; }
  v8f acc[2] = {};
  for (int c0 = 0; c0 < CCH; c0 += 32) {
    _Float16 fv[8], gv[8];
#pragma unroll
    for (int i = 0; i < 8; ++i) {
      int k = c0 + kbase + i;
      fv[i] = F[k * MTOT + gmL];
      gv[i] = G[k * MTOT + gmM];
    }
#pragma unroll
    for (int i = 0; i < 8; ++i) {
      Fs[rloc * LPAD + kbase + i] = fv[i];
      Gs[rloc * LPAD + kbase + i] = gv[i];
    }
    __syncthreads();
    v16h bf = load_frag(&Gs[(waveM * 16 + (lane & 15)) * LPAD], lane);
#pragma unroll
    for (int li = 0; li < 2; ++li) {
      v16h af = load_frag(&Fs[(waveL * 32 + li * 16 + (lane & 15)) * LPAD], lane);
      acc[li] = __builtin_amdgcn_wmma_f32_16x16x32_f16(false, af, false, bf, (short)0,
                                                       acc[li], false, false);
    }
    __syncthreads();
  }
  const int rowOff = (lane & 16) ? 8 : 0;
  const int mcol = mBase + waveM * 16 + (lane & 15);
  const int mok = (mcol < Lp);
#pragma unroll
  for (int li = 0; li < 2; ++li)
#pragma unroll
    for (int r = 0; r < 8; ++r) {
      int ll = waveL * 32 + li * 16 + rowOff + r;            // row within tile
      float v = (mok && (lBase + ll) < Lp) ? acc[li][r] : 0.f;  // zero the pad region
      Sblk[ll * Lpd + mcol] = v;
    }
}

// ---------------- row softmax over first Lp cols (pad stays zero) -----------------
__global__ void softmax_kernel(float* S, int Lp, int Lpd) {
  int n = blockIdx.x / Lp, l = blockIdx.x % Lp;
  float* rp = S + (n * Lpd + l) * Lpd;
  int tid = threadIdx.x;
  __shared__ float red[256];
  float mx = -1e30f;
  for (int i = tid; i < Lp; i += 256) mx = fmaxf(mx, rp[i]);
  red[tid] = mx; __syncthreads();
  for (int s = 128; s > 0; s >>= 1) { if (tid < s) red[tid] = fmaxf(red[tid], red[tid + s]); __syncthreads(); }
  mx = red[0]; __syncthreads();
  float sm = 0.f;
  for (int i = tid; i < Lp; i += 256) { float e = expf(rp[i] - mx); rp[i] = e; sm += e; }
  red[tid] = sm; __syncthreads();
  for (int s = 128; s > 0; s >>= 1) { if (tid < s) red[tid] += red[tid + s]; __syncthreads(); }
  float inv = 1.f / red[0];
  for (int i = tid; i < Lp; i += 256) rp[i] *= inv;
}

// ---------------- attention output: O[c,l] = sum_m H[c,m]*S[l,m] ------------------
// Fully unguarded staging: S pad (rows & cols >= Lp) is zero, H uses clamped index.
__global__ __launch_bounds__(256) void attn_out_kernel(const _Float16* H, const float* S,
                                                       _Float16* O, int Lp, int Lpd,
                                                       int plen, int offsj) {
  const int n = blockIdx.z;
  const int cBase = blockIdx.y * 64, lBase = blockIdx.x * 64;
  const float* Sblk = S + (n * Lpd + lBase) * Lpd;  // block-uniform base, 32-bit offsets
  __shared__ _Float16 Hs[64 * LPAD];
  __shared__ _Float16 Ss[64 * LPAD];
  const int tid = threadIdx.x, lane = tid & 31, wave = tid >> 5;
  const int waveC = wave >> 2, waveL = wave & 3;   // wave: 32(c) x 16(l)
  v8f acc[2] = {};
  for (int m0 = 0; m0 < Lp; m0 += 32) {
    // K axis (m) is the contiguous axis of both H and S: lanes walk k.
    int mm = m0 + lane;
    int mmc = (mm < Lp) ? mm : Lp - 1;             // clamp; S pad is zero so product = 0
    int gmm;
    { int t = mmc / plen, j = mmc - t * plen; gmm = (n * TT + t) * VV + offsj + j; }
    _Float16 hv[8], sv[8];
#pragma unroll
    for (int i = 0; i < 8; ++i) {
      int r = wave + 8 * i;
      hv[i] = H[(cBase + r) * MTOT + gmm];
      sv[i] = (_Float16)Sblk[r * Lpd + mm];
    }
#pragma unroll
    for (int i = 0; i < 8; ++i) {
      int r = wave + 8 * i;
      Hs[r * LPAD + lane] = hv[i];
      Ss[r * LPAD + lane] = sv[i];
    }
    __syncthreads();
    v16h bf = load_frag(&Ss[(waveL * 16 + (lane & 15)) * LPAD], lane);
#pragma unroll
    for (int ci = 0; ci < 2; ++ci) {
      v16h af = load_frag(&Hs[(waveC * 32 + ci * 16 + (lane & 15)) * LPAD], lane);
      acc[ci] = __builtin_amdgcn_wmma_f32_16x16x32_f16(false, af, false, bf, (short)0,
                                                       acc[ci], false, false);
    }
    __syncthreads();
  }
  const int rowOff = (lane & 16) ? 8 : 0;
  const int lcol = lBase + waveL * 16 + (lane & 15);
  if (lcol < Lp) {
    int t = lcol / plen, j = lcol - t * plen;
    int gm = (n * TT + t) * VV + offsj + j;
#pragma unroll
    for (int ci = 0; ci < 2; ++ci)
#pragma unroll
      for (int r = 0; r < 8; ++r) {
        int c = cBase + waveC * 32 + ci * 16 + rowOff + r;
        O[c * MTOT + gm] = (_Float16)acc[ci][r];
      }
  }
}

// ================================ host launcher ===================================
extern "C" void kernel_launch(void* const* d_in, const int* in_sizes, int n_in,
                              void* d_out, int out_size, void* d_ws, size_t ws_size,
                              hipStream_t stream) {
  (void)in_sizes; (void)n_in; (void)out_size; (void)ws_size;
  const float* x = (const float*)d_in[0];
  const float* sIn[5] = {(const float*)d_in[1], (const float*)d_in[2], (const float*)d_in[3],
                         (const float*)d_in[4], (const float*)d_in[5]};
  const float* Amat = (const float*)d_in[6];
  const float* awl = (const float*)d_in[7],  *abl = (const float*)d_in[8];
  const float* aw1 = (const float*)d_in[9],  *ab1 = (const float*)d_in[10];
  const float* aw2 = (const float*)d_in[11], *ab2 = (const float*)d_in[12];
  const float* gcn1_w = (const float*)d_in[13], *gcn1_b = (const float*)d_in[14];
  const float* tcn1_w = (const float*)d_in[15], *tcn1_b = (const float*)d_in[16];
  const float* ast_wf = (const float*)d_in[17], *ast_bf = (const float*)d_in[18];
  const float* ast_wg = (const float*)d_in[19], *ast_bg = (const float*)d_in[20];
  const float* ast_wh = (const float*)d_in[21], *ast_bh = (const float*)d_in[22];
  const float* ast_wk = (const float*)d_in[23], *ast_bk = (const float*)d_in[24];
  const float* gcn2_w = (const float*)d_in[25], *gcn2_b = (const float*)d_in[26];
  const float* tcn2_w = (const float*)d_in[27], *tcn2_b = (const float*)d_in[28];
  float* outp = (float*)d_out;

  // ---- workspace carve ----
  char* wp = (char*)d_ws;
  auto carve = [&](size_t b) { char* r = wp; wp += (b + 255) & ~(size_t)255; return (void*)r; };
  const size_t cmF32 = (size_t)CCH * MTOT * sizeof(float);
  const size_t cmF16 = (size_t)CCH * MTOT * sizeof(_Float16);
  float*    act3  = (float*)carve(cmF32);                 // tcn1 out (astyle x, residual)
  _Float16* act1h = (_Float16*)carve(cmF16);              // adain out; reused as act4h
  _Float16* act2h = (_Float16*)carve(cmF16);              // gcn-contract out; reused as act5h
  _Float16* Yb    = (_Float16*)carve(3 * cmF16);          // gemm Y; aliased by Sb (26MB < 124MB)
  _Float16* Xn    = (_Float16*)carve(cmF16);              // inorm(act3) f16
  _Float16* Sn    = (_Float16*)carve(cmF16);              // inorm(style) f16
  _Float16* Sr    = (_Float16*)carve(cmF16);              // raw style f16
  _Float16* Fb    = (_Float16*)carve(cmF16);              // F out; reused as Ob
  _Float16* Gb    = (_Float16*)carve(cmF16);              // G out
  _Float16* Hb    = (_Float16*)carve(cmF16);              // H out
  float* sMean = (float*)carve(5 * NB * CCH * sizeof(float));
  float* sRstd = (float*)carve(5 * NB * CCH * sizeof(float));
  float* xMean = (float*)carve(5 * NB * CCH * sizeof(float));
  float* xRstd = (float*)carve(5 * NB * CCH * sizeof(float));
  float* fMean = (float*)carve(5 * NB * CCH * sizeof(float));
  float* fRstd = (float*)carve(5 * NB * CCH * sizeof(float));
  float* gbeta = (float*)carve(5 * NB * 2 * CCH * sizeof(float));
  float*    Sb    = (float*)Yb;     // alias: Y dead during attention
  _Float16* act4h = act1h;          // alias
  _Float16* act5h = act2h;          // alias
  _Float16* Ob    = Fb;             // alias: F consumed by scores before O is written

  static const int offH[5] = {0, 4, 8, 13, 17};
  static const int lenH[5] = {4, 4, 5, 4, 4};
  dim3 blk(256);
  auto gemm = [&](auto kern, GemmArgs g) {
    dim3 grid(g.Mdim / 128, g.Odim / 64);
    kern<<<grid, blk, 0, stream>>>(g);
  };

  // ---- stage 0: stats + AdaIN ----
  for (int p = 0; p < 5; ++p) {
    stats_nctv_kernel<<<dim3(NB), blk, 0, stream>>>(sIn[p], p, sMean, sRstd);
    stats_nctv_kernel<<<dim3(NB), blk, 0, stream>>>(x, p, xMean, xRstd);
  }
  adain_mlp_kernel<<<dim3(NB, 5), blk, 0, stream>>>(sMean, awl, abl, aw1, ab1, aw2, ab2, gbeta);
  adain_apply_kernel<<<dim3(MTOT), blk, 0, stream>>>(x, xMean, xRstd, gbeta, act1h);
  pack_style_kernel<<<dim3(MTOT), blk, 0, stream>>>(sIn[0], sIn[1], sIn[2], sIn[3], sIn[4],
                                                    sMean, sRstd, Sn, Sr);

  // ---- stage 1: GCN1 + A-contraction + TCN1 ----
  { GemmArgs g = {}; g.W = gcn1_w; g.B = act1h; g.out16 = Yb;
    g.Odim = 3 * CCH; g.Kdim = CCH; g.Mdim = MTOT; g.plen = VV;
    gemm(gemm_wmma_kernel<0, 2, 1>, g); }
  acontract_kernel<<<dim3(NTT), blk, 0, stream>>>(Yb, Amat, gcn1_b, act2h);
  { GemmArgs g = {}; g.W = tcn1_w; g.bias = tcn1_b; g.B = act2h; g.out32 = act3;
    g.Odim = CCH; g.Kdim = 3 * CCH; g.Mdim = MTOT; g.plen = VV;
    gemm(gemm_wmma_kernel<1, 3, 0>, g); }

  // ---- stage 2: per-part attention (astyle) ----
  for (int p = 0; p < 5; ++p)
    stats_cm_kernel<<<dim3(NB), blk, 0, stream>>>(act3, p, fMean, fRstd);
  normalize_cm_kernel<<<dim3(MTOT), blk, 0, stream>>>(act3, fMean, fRstd, Xn);

  for (int p = 0; p < 5; ++p) {
    const int len = lenH[p], off = offH[p];
    const int Lp = TT * len, Mp = NB * TT * len;
    const int lt = (Lp + 63) / 64, Lpd = lt * 64;
    { GemmArgs g = {}; g.W = ast_wf + (size_t)p * CCH * CCH; g.bias = ast_bf + p * CCH;
      g.B = Xn; g.out16 = Fb; g.Odim = CCH; g.Kdim = CCH; g.Mdim = Mp;
      g.offs = off; g.plen = len; gemm(gemm_wmma_kernel<0, 2, 1>, g); }
    { GemmArgs g = {}; g.W = ast_wg + (size_t)p * CCH * CCH; g.bias = ast_bg + p * CCH;
      g.B = Sn; g.out16 = Gb; g.Odim = CCH; g.Kdim = CCH; g.Mdim = Mp;
      g.offs = off; g.plen = len; gemm(gemm_wmma_kernel<0, 2, 1>, g); }
    { GemmArgs g = {}; g.W = ast_wh + (size_t)p * CCH * CCH; g.bias = ast_bh + p * CCH;
      g.B = Sr; g.out16 = Hb; g.Odim = CCH; g.Kdim = CCH; g.Mdim = Mp;
      g.offs = off; g.plen = len; gemm(gemm_wmma_kernel<0, 2, 1>, g); }
    attn_scores_kernel<<<dim3(lt, lt, NB), blk, 0, stream>>>(Fb, Gb, Sb, Lp, Lpd, len, off);
    softmax_kernel<<<dim3(NB * Lp), blk, 0, stream>>>(Sb, Lp, Lpd);
    attn_out_kernel<<<dim3(lt, 4, NB), blk, 0, stream>>>(Hb, Sb, Ob, Lp, Lpd, len, off);
    { GemmArgs g = {}; g.W = ast_wk + (size_t)p * CCH * CCH; g.bias = ast_bk + p * CCH;
      g.B = Ob; g.resid = act3; g.out16 = act4h; g.Odim = CCH; g.Kdim = CCH; g.Mdim = Mp;
      g.offs = off; g.plen = len; gemm(gemm_wmma_kernel<0, 2, 1>, g); }
  }

  // ---- stage 3: GCN2 + A-contraction + TCN2 -> d_out ----
  { GemmArgs g = {}; g.W = gcn2_w; g.B = act4h; g.out16 = Yb;
    g.Odim = 3 * CCH; g.Kdim = CCH; g.Mdim = MTOT; g.plen = VV;
    gemm(gemm_wmma_kernel<0, 2, 1>, g); }
  acontract_kernel<<<dim3(NTT), blk, 0, stream>>>(Yb, Amat, gcn2_b, act5h);
  { GemmArgs g = {}; g.W = tcn2_w; g.bias = tcn2_b; g.B = act5h; g.outNC = outp;
    g.Odim = CCH; g.Kdim = 3 * CCH; g.Mdim = MTOT; g.plen = VV;
    gemm(gemm_wmma_kernel<1, 3, 2>, g); }
}
